// ScreeningAttentionTriton_15290083574090
// MI455X (gfx1250) — matrix operand order repaired and checked by reading, not compiled, not run on hardware
//
#include <hip/hip_runtime.h>
#include <hip/hip_bf16.h>

typedef __attribute__((ext_vector_type(16))) __bf16 v16bf;
typedef __attribute__((ext_vector_type(8)))  __bf16 v8bf;
typedef __attribute__((ext_vector_type(8)))  float  v8f;

union V16U { v16bf v; v8bf h[2]; };

static __device__ __forceinline__ unsigned short f2bf(float f) {
  __bf16 h = (__bf16)f;
  return __builtin_bit_cast(unsigned short, h);
}
static __device__ __forceinline__ unsigned pk2(float a, float b) {
  return (unsigned)f2bf(a) | ((unsigned)f2bf(b) << 16);
}
static __device__ __forceinline__ v8f wmma_bf16(v16bf a, v16bf b, v8f c) {
  // (neg_a, A, neg_b, B, c_mod, C, reuse_a, reuse_b)
  return __builtin_amdgcn_wmma_f32_16x16x32_bf16(false, a, false, b, (short)0, c, false, false);
}

// Problem constants (reference: B=2, T=4096, Dm=1024, H=16, D=64, W_BLK=256)
#define TT   4096
#define DM   1024
#define HH   16
#define DD   64
#define WBL  256

// MODE 0: f32 input, l2norm epilogue, out bf16 [B,H,T,D]
// MODE 1: f32 input, l2norm epilogue, out bf16 [B,H,D,T] (transposed, for V)
// MODE 2: bf16 input, plain f32 out [rows, DM]
template <int MODE>
__global__ __launch_bounds__(256, 1)
void gemm_xwt(const void* __restrict__ Xv, const float* __restrict__ W,
              void* __restrict__ Out) {
  extern __shared__ char smem[];
  // double-buffered tiles: [2][128][48] X, [2][128][48] W, then f32 epilogue staging
  unsigned short* Xs = (unsigned short*)smem;       // buf b at Xs + b*128*48
  unsigned short* Ws = Xs + 2 * 128 * 48;           // buf b at Ws + b*128*48
  float*          Cs = (float*)(Ws + 2 * 128 * 48); // [128][132]

  const int tid  = threadIdx.x;
  const int lane = tid & 31;
  const int wv   = tid >> 5;
  const int l15  = lane & 15;
  const int lh   = lane >> 4;
  const int m0   = blockIdx.x * 128;
  const int n0   = blockIdx.y * 128;
  const int wr   = (wv & 3) * 32;   // wave's row offset inside tile
  const int wc   = (wv >> 2) * 64;  // wave's col offset inside tile (one head wide)

  v8f acc[2][4];
#pragma unroll
  for (int i = 0; i < 2; i++)
#pragma unroll
    for (int j = 0; j < 4; j++) acc[i][j] = {};

  const int sr = tid >> 1;         // staging row 0..127
  const int sc = (tid & 1) * 16;   // staging col segment
  const int NK = DM / 32;

  // ---- prologue: stage tile 0 into buffer 0 ----
  if (MODE < 2) {
    const float* X = (const float*)Xv;
    const float4* px = (const float4*)(X + (size_t)(m0 + sr) * DM + sc);
    float4 f0 = px[0], f1 = px[1], f2 = px[2], f3 = px[3];
    uint4 u0 = {pk2(f0.x, f0.y), pk2(f0.z, f0.w), pk2(f1.x, f1.y), pk2(f1.z, f1.w)};
    uint4 u1 = {pk2(f2.x, f2.y), pk2(f2.z, f2.w), pk2(f3.x, f3.y), pk2(f3.z, f3.w)};
    uint4* dx = (uint4*)(Xs + sr * 48 + sc);
    dx[0] = u0; dx[1] = u1;
  } else {
    const unsigned short* X = (const unsigned short*)Xv;
    const uint4* px = (const uint4*)(X + (size_t)(m0 + sr) * DM + sc);
    uint4* dx = (uint4*)(Xs + sr * 48 + sc);
    dx[0] = px[0]; dx[1] = px[1];
  }
  {
    const float4* pw = (const float4*)(W + (size_t)(n0 + sr) * DM + sc);
    float4 f0 = pw[0], f1 = pw[1], f2 = pw[2], f3 = pw[3];
    uint4 u0 = {pk2(f0.x, f0.y), pk2(f0.z, f0.w), pk2(f1.x, f1.y), pk2(f1.z, f1.w)};
    uint4 u1 = {pk2(f2.x, f2.y), pk2(f2.z, f2.w), pk2(f3.x, f3.y), pk2(f3.z, f3.w)};
    uint4* dw = (uint4*)(Ws + sr * 48 + sc);
    dw[0] = u0; dw[1] = u1;
  }
  __syncthreads();

#pragma unroll 1
  for (int kt = 0; kt < NK; kt++) {
    const int cur = kt & 1;
    const int nxt = cur ^ 1;
    const bool have = (kt + 1 < NK);
    const int kk2 = (kt + 1) * 32;

    // ---- issue next tile's global loads into registers (latency overlapped) ----
    float4 xf0, xf1, xf2, xf3, wf0, wf1, wf2, wf3;
    uint4 xu0, xu1;
    if (have) {
      if (MODE < 2) {
        const float* X = (const float*)Xv;
        const float4* px = (const float4*)(X + (size_t)(m0 + sr) * DM + kk2 + sc);
        xf0 = px[0]; xf1 = px[1]; xf2 = px[2]; xf3 = px[3];
      } else {
        const unsigned short* X = (const unsigned short*)Xv;
        const uint4* px = (const uint4*)(X + (size_t)(m0 + sr) * DM + kk2 + sc);
        xu0 = px[0]; xu1 = px[1];
      }
      const float4* pw = (const float4*)(W + (size_t)(n0 + sr) * DM + kk2 + sc);
      wf0 = pw[0]; wf1 = pw[1]; wf2 = pw[2]; wf3 = pw[3];
    }

    // ---- compute current tile: 8 WMMAs per wave ----
    const unsigned short* Xc = Xs + cur * 128 * 48;
    const unsigned short* Wc = Ws + cur * 128 * 48;
    v16bf afrag[2];
#pragma unroll
    for (int mt = 0; mt < 2; mt++) {
      const unsigned short* pa = Xc + (wr + mt * 16 + l15) * 48 + lh * 8;
      V16U u; u.h[0] = *(const v8bf*)pa; u.h[1] = *(const v8bf*)(pa + 16);
      afrag[mt] = u.v;
    }
#pragma unroll
    for (int nt = 0; nt < 4; nt++) {
      const unsigned short* pb = Wc + (wc + nt * 16 + l15) * 48 + lh * 8;
      V16U ub; ub.h[0] = *(const v8bf*)pb; ub.h[1] = *(const v8bf*)(pb + 16);
      v16bf bfrag = ub.v;
#pragma unroll
      for (int mt = 0; mt < 2; mt++) acc[mt][nt] = wmma_bf16(afrag[mt], bfrag, acc[mt][nt]);
    }

    // ---- store next tile into the alternate buffer ----
    if (have) {
      unsigned short* Xn = Xs + nxt * 128 * 48;
      unsigned short* Wn = Ws + nxt * 128 * 48;
      if (MODE < 2) {
        uint4 u0 = {pk2(xf0.x, xf0.y), pk2(xf0.z, xf0.w), pk2(xf1.x, xf1.y), pk2(xf1.z, xf1.w)};
        uint4 u1 = {pk2(xf2.x, xf2.y), pk2(xf2.z, xf2.w), pk2(xf3.x, xf3.y), pk2(xf3.z, xf3.w)};
        uint4* dx = (uint4*)(Xn + sr * 48 + sc);
        dx[0] = u0; dx[1] = u1;
      } else {
        uint4* dx = (uint4*)(Xn + sr * 48 + sc);
        dx[0] = xu0; dx[1] = xu1;
      }
      uint4 w0 = {pk2(wf0.x, wf0.y), pk2(wf0.z, wf0.w), pk2(wf1.x, wf1.y), pk2(wf1.z, wf1.w)};
      uint4 w1 = {pk2(wf2.x, wf2.y), pk2(wf2.z, wf2.w), pk2(wf3.x, wf3.y), pk2(wf3.z, wf3.w)};
      uint4* dw = (uint4*)(Wn + sr * 48 + sc);
      dw[0] = w0; dw[1] = w1;
    }
    __syncthreads();
  }

  // ---- spill accumulators to LDS f32 for cross-lane epilogue ----
#pragma unroll
  for (int mt = 0; mt < 2; mt++)
#pragma unroll
    for (int nt = 0; nt < 4; nt++)
#pragma unroll
      for (int r = 0; r < 8; r++)
        Cs[(wr + mt * 16 + lh * 8 + r) * 132 + wc + nt * 16 + l15] = acc[mt][nt][r];
  __syncthreads();

  // ---- epilogue: 2 threads per row, each owns one 64-wide head slice ----
  const int er = tid >> 1, eh = tid & 1;
  const int gi = m0 + er;                 // flattened (b,t) row
  if (MODE == 2) {
    float* Outf = (float*)Out;
#pragma unroll 8
    for (int c = 0; c < 64; c++)
      Outf[(size_t)gi * DM + n0 + eh * 64 + c] = Cs[er * 132 + eh * 64 + c];
  } else {
    float s = 0.f;
#pragma unroll 8
    for (int c = 0; c < 64; c++) { float x = Cs[er * 132 + eh * 64 + c]; s += x * x; }
    float inv = 1.f / fmaxf(sqrtf(s), 1e-8f);
    const int bb = gi >> 12;              // / T
    const int trow = gi & (TT - 1);
    const int hd = (n0 >> 6) + eh;        // head index
    unsigned short* Ob = (unsigned short*)Out;
    if (MODE == 0) {
      unsigned short* dst = Ob + ((size_t)(bb * HH + hd) * TT + trow) * DD;
#pragma unroll 8
      for (int c = 0; c < 64; c++) dst[c] = f2bf(Cs[er * 132 + eh * 64 + c] * inv);
    } else {
      unsigned short* dst = Ob + ((size_t)(bb * HH + hd) * DD) * TT + trow;
#pragma unroll 8
      for (int c = 0; c < 64; c++) dst[(size_t)c * TT] = f2bf(Cs[er * 132 + eh * 64 + c] * inv);
    }
  }
}

// One workgroup per (block n, head h, batch b). 256 threads = 8 waves.
__global__ __launch_bounds__(256, 1)
void screen_attn(const unsigned short* __restrict__ Qb,
                 const unsigned short* __restrict__ Kb,
                 const unsigned short* __restrict__ VT,
                 const float* __restrict__ svec,
                 const float* __restrict__ rvec,
                 unsigned short* __restrict__ Ab) {
  extern __shared__ char smem[];
  unsigned short* sq  = (unsigned short*)smem; // [256][64]  q block
  unsigned short* sk  = sq + 256 * 64;         // [512][64]  k prev||cur
  unsigned short* svT = sk + 512 * 64;         // [64][512]  v transposed
  unsigned short* sal = svT + 64 * 512;        // [256][128] alpha chunk (bf16) / reused f32[256][64]
  float*          lut = (float*)(sal + 256 * 128); // [768] cosine window LUT, idx = rel + 511

  const int tid  = threadIdx.x;
  const int lane = tid & 31;
  const int wv   = tid >> 5;
  const int l15  = lane & 15;
  const int lh   = lane >> 4;
  const int n = (int)blockIdx.x, h = (int)blockIdx.y, b = (int)blockIdx.z;

  const float wH = __expf(svec[h]) + 1.0f;
  const float rH = __expf(rvec[h]) + 1.0f;
  const float oneMinusR = 1.0f - rH;
  const float piOverW = 3.14159265358979f / wH;   // one divide per thread, hoisted

  // ---- build mask LUT: rel in (-512, 256); lut[idx] = window(rel = idx - 511) ----
#pragma unroll
  for (int t = 0; t < 3; t++) {
    int idx = tid + t * 256;
    float relf = (float)(idx - 511);
    float v = 0.0f;
    if (relf > -wH && relf <= 0.0f) v = 0.5f * (__cosf(piOverW * relf) + 1.0f);
    lut[idx] = v;
  }

  // ---- stage q: row per thread (128B = 8x uint4) ----
  {
    const uint4* src = (const uint4*)(Qb + ((size_t)((b * HH + h) * TT) + n * WBL + tid) * DD);
    uint4* dst = (uint4*)(sq + tid * DD);
#pragma unroll
    for (int u = 0; u < 8; u++) dst[u] = src[u];
  }
  // ---- stage kc: 512 rows = prev block || cur block (prev of block 0 is zero) ----
#pragma unroll
  for (int rr = 0; rr < 2; rr++) {
    int j = 2 * tid + rr;
    long tsrc = (long)(n - 1) * WBL + j;
    uint4* dst = (uint4*)(sk + j * DD);
    if (tsrc >= 0) {
      const uint4* src = (const uint4*)(Kb + ((size_t)((b * HH + h) * TT) + tsrc) * DD);
#pragma unroll
      for (int u = 0; u < 8; u++) dst[u] = src[u];
    } else {
      uint4 z = {0, 0, 0, 0};
#pragma unroll
      for (int u = 0; u < 8; u++) dst[u] = z;
    }
  }
  // ---- stage vT: 64 rows x 512 cols; thread covers 128 contiguous cols of one row ----
  {
    int d = tid >> 2, q4 = tid & 3;
    long c0 = (long)(n - 1) * WBL + q4 * 128;
    uint4* dst = (uint4*)(svT + d * 512 + q4 * 128);
    if (c0 >= 0) {
      const uint4* src = (const uint4*)(VT + ((size_t)((b * HH + h) * DD + d)) * TT + c0);
#pragma unroll
      for (int u = 0; u < 16; u++) dst[u] = src[u];
    } else {
      uint4 z = {0, 0, 0, 0};
#pragma unroll
      for (int u = 0; u < 16; u++) dst[u] = z;
    }
  }
  __syncthreads();

  const int mrow = wv * 32;  // wave owns 32 rows of the 256-row block

  // ---- hoist q A-fragments: invariant over nt AND chunk (sq never overwritten) ----
  v16bf aq[2][2];  // [mt][ks]
#pragma unroll
  for (int mt = 0; mt < 2; mt++)
#pragma unroll
    for (int ks = 0; ks < 2; ks++) {
      const unsigned short* pa = sq + (mrow + mt * 16 + l15) * DD + ks * 32 + lh * 8;
      V16U u; u.h[0] = *(const v8bf*)pa; u.h[1] = *(const v8bf*)(pa + 16);
      aq[mt][ks] = u.v;
    }

  v8f o[2][4];
#pragma unroll
  for (int i = 0; i < 2; i++)
#pragma unroll
    for (int j = 0; j < 4; j++) o[i][j] = {};

#pragma unroll 1
  for (int ch = 0; ch < 4; ch++) {
    // ---- GEMM1: sim[256 x 128] for this column chunk, mask -> alpha in LDS ----
#pragma unroll 1
    for (int nt = 0; nt < 8; nt++) {
      int jc = ch * 128 + nt * 16;  // global kc column base
      v16bf bk[2];
#pragma unroll
      for (int ks = 0; ks < 2; ks++) {
        const unsigned short* pb = sk + (jc + l15) * DD + ks * 32 + lh * 8;
        V16U ub; ub.h[0] = *(const v8bf*)pb; ub.h[1] = *(const v8bf*)(pb + 16);
        bk[ks] = ub.v;
      }
#pragma unroll
      for (int mt = 0; mt < 2; mt++) {
        v8f acc = {};
#pragma unroll
        for (int ks = 0; ks < 2; ks++) acc = wmma_bf16(aq[mt][ks], bk[ks], acc);
        const int jcol = jc + l15;
        const float bval = (n > 0 || jcol >= WBL) ? 1.0f : 0.0f;
        const int irow0 = mrow + mt * 16 + lh * 8;       // row of acc[0]
        const int bidx = jcol - irow0 + 511 - WBL;       // lut index for r = 0
        unsigned short* sp = sal + irow0 * 128 + nt * 16 + l15;
#pragma unroll
        for (int r = 0; r < 8; r++) {
          float m = lut[bidx - r] * bval;
          float a = fmaxf(fmaf(rH, acc[r], oneMinusR), 0.0f);
          sp[r * 128] = f2bf(a * a * m);
        }
      }
    }
    __syncthreads();
    // ---- GEMM2: o += alpha_chunk @ v_chunk (K = 128) ----
#pragma unroll 1
    for (int ks = 0; ks < 4; ks++) {
      v16bf af[2];
#pragma unroll
      for (int mt = 0; mt < 2; mt++) {
        const unsigned short* pa = sal + (mrow + mt * 16 + l15) * 128 + ks * 32 + lh * 8;
        V16U u; u.h[0] = *(const v8bf*)pa; u.h[1] = *(const v8bf*)(pa + 16);
        af[mt] = u.v;
      }
#pragma unroll
      for (int nt = 0; nt < 4; nt++) {
        const unsigned short* pb = svT + (nt * 16 + l15) * 512 + ch * 128 + ks * 32 + lh * 8;
        V16U ub; ub.h[0] = *(const v8bf*)pb; ub.h[1] = *(const v8bf*)(pb + 16);
        v16bf bfr = ub.v;
#pragma unroll
        for (int mt = 0; mt < 2; mt++) o[mt][nt] = wmma_bf16(af[mt], bfr, o[mt][nt]);
      }
    }
    __syncthreads();
  }

  // ---- epilogue: tanh(||out||)/||out|| per row, store bf16 to [B,T,Dm] ----
  float* sf = (float*)sal;  // reuse as f32 [256][64]
#pragma unroll
  for (int mt = 0; mt < 2; mt++)
#pragma unroll
    for (int nt = 0; nt < 4; nt++)
#pragma unroll
      for (int r = 0; r < 8; r++)
        sf[(mrow + mt * 16 + lh * 8 + r) * 64 + nt * 16 + l15] = o[mt][nt][r];
  __syncthreads();
  {
    float s = 0.f;
#pragma unroll 8
    for (int c = 0; c < 64; c++) { float x = sf[tid * 64 + c]; s += x * x; }
    float nn = sqrtf(s);
    float sc = tanhf(nn) / (nn + 1e-8f);
    unsigned short* dst = Ab + ((size_t)(b * TT + n * WBL + tid)) * DM + h * DD;
#pragma unroll 8
    for (int c = 0; c < 64; c++) dst[c] = f2bf(sf[tid * 64 + c] * sc);
  }
}

extern "C" void kernel_launch(void* const* d_in, const int* in_sizes, int n_in,
                              void* d_out, int out_size, void* d_ws, size_t ws_size,
                              hipStream_t stream) {
  (void)in_sizes; (void)n_in; (void)out_size; (void)ws_size;
  const float* query = (const float*)d_in[0];
  const float* key   = (const float*)d_in[1];
  const float* value = (const float*)d_in[2];
  const float* Wq    = (const float*)d_in[3];
  const float* Wk    = (const float*)d_in[4];
  const float* Wv    = (const float*)d_in[5];
  const float* Wo    = (const float*)d_in[6];
  const float* s_v   = (const float*)d_in[7];
  const float* s_r   = (const float*)d_in[8];

  char* ws = (char*)d_ws;
  const size_t SZ = (size_t)2 * HH * TT * DD * 2;  // 16 MB per bf16 tensor
  unsigned short* qbuf  = (unsigned short*)ws;
  unsigned short* kbuf  = (unsigned short*)(ws + SZ);
  unsigned short* vbufT = (unsigned short*)(ws + 2 * SZ);
  unsigned short* abuf  = (unsigned short*)(ws + 3 * SZ);

  const size_t smemG = (size_t)2 * 128 * 48 * 2 * 2 + (size_t)128 * 132 * 4;  // 116736 B
  dim3 gG(64, 8);
  gemm_xwt<0><<<gG, 256, smemG, stream>>>(query, Wq, qbuf);
  gemm_xwt<0><<<gG, 256, smemG, stream>>>(key,   Wk, kbuf);
  gemm_xwt<1><<<gG, 256, smemG, stream>>>(value, Wv, vbufT);

  const size_t smemA = (size_t)(256 * 64 + 512 * 64 + 64 * 512 + 256 * 128) * 2 + 768 * 4;
  dim3 gA(16, 16, 2);
  screen_attn<<<gA, 256, smemA, stream>>>(qbuf, kbuf, vbufT, s_v, s_r, abuf);

  gemm_xwt<2><<<gG, 256, smemG, stream>>>(abuf, Wo, (float*)d_out);
}
